// FeatureExtractor_80161269612858
// MI455X (gfx1250) — matrix operand order repaired and checked by reading, not compiled, not run on hardware
//
#include <hip/hip_runtime.h>

// Problem constants (fixed by the reference setup_inputs()).
#define BQ  16384   // batch
#define FQ  1024    // features
#define DQ  4       // ensemble size
#define O2Q 32      // second-layer output width
#define EPSQ 1e-5f

// ---------------------------------------------------------------------------
// CDNA5 WMMA types
// ---------------------------------------------------------------------------
typedef __attribute__((ext_vector_type(16))) __bf16 v16bf;
typedef __attribute__((ext_vector_type(8)))  float  v8f;

struct alignas(16) Chunk16B { unsigned int w[4]; };   // POD 16-byte load unit
union Frag32B { v16bf v; Chunk16B q[2]; };

__device__ __forceinline__ unsigned short f2bf(float x) {
  unsigned u = __builtin_bit_cast(unsigned, x);
  u += 0x7FFFu + ((u >> 16) & 1u);               // round-to-nearest-even
  return (unsigned short)(u >> 16);
}
__device__ __forceinline__ float bf2f(unsigned short h) {
  unsigned u = ((unsigned)h) << 16;
  return __builtin_bit_cast(float, u);
}
__device__ __forceinline__ float sigmoidf_(float x) {
  return 1.0f / (1.0f + __expf(-x));
}

// A fragment: 16x32 bf16 (MxK), row-major source with leading dim lda.
// Lane m = lane&15 ; half = lane>>4.
// VGPRs 0-3 hold K = half*8 + 0..7 (contiguous), VGPRs 4-7 hold K = 16+half*8+0..7.
__device__ __forceinline__ v16bf load_a_frag(const unsigned short* __restrict__ A,
                                             int lda, int row, int k0, int half) {
  Frag32B f;
  const unsigned short* p = A + (size_t)row * lda + k0 + half * 8;
  f.q[0] = *(const Chunk16B*)(p);
  f.q[1] = *(const Chunk16B*)(p + 16);
  return f.v;
}

// B fragment: 32x16 bf16 (KxN); W stored row-major [N, K] (i.e. B(k,n)=W[n,k]).
// Lane n = lane&15 ; half = lane>>4.
// VGPRs 0-3 hold K = half*8 + 0..7, VGPRs 4-7 hold K = 16+half*8+0..7 (matches A).
__device__ __forceinline__ v16bf load_b_frag(const unsigned short* __restrict__ W,
                                             int ldw, int col, int k0, int half) {
  Frag32B f;
  const unsigned short* p = W + (size_t)col * ldw + k0 + half * 8;
  f.q[0] = *(const Chunk16B*)(p);
  f.q[1] = *(const Chunk16B*)(p + 16);
  return f.v;
}

// ---------------------------------------------------------------------------
// WMMA GEMM: out[d][m,n] = sum_k A[d][m,k] * W[d][n,k] (+ bias[d][n])
// Block = 128 threads (4 waves). Each wave computes a 32x32 tile (2x2 WMMAs).
// Grid: (M/128, N/32, D).
// ---------------------------------------------------------------------------
template <bool OUT_BF16>
__global__ __launch_bounds__(128)
void gemm_wmma_bf16(const unsigned short* __restrict__ A,
                    const unsigned short* __restrict__ W,
                    const float* __restrict__ bias,
                    void* __restrict__ Out,
                    int M, int N, int K,
                    size_t sAd, size_t sWd, size_t sOd, size_t sBd) {
  const int d = blockIdx.z;
  const unsigned short* Ad = A + (size_t)d * sAd;
  const unsigned short* Wd = W + (size_t)d * sWd;

  const int lane   = threadIdx.x & 31;
  const int wave   = threadIdx.x >> 5;
  const int half   = lane >> 4;
  const int lane15 = lane & 15;

  const int m0 = blockIdx.x * 128 + wave * 32;
  const int n0 = blockIdx.y * 32;

  const int rowA0 = m0 + lane15;
  const int rowA1 = m0 + 16 + lane15;
  const int colB0 = n0 + lane15;
  const int colB1 = n0 + 16 + lane15;

  v8f acc00 = {}; v8f acc01 = {}; v8f acc10 = {}; v8f acc11 = {};

  for (int k0 = 0; k0 < K; k0 += 32) {
    if (k0 + 32 < K) {   // warm L2/L0 for next K-tile -> global_prefetch_b8
      __builtin_prefetch(Ad + (size_t)rowA0 * K + k0 + 32, 0, 3);
      __builtin_prefetch(Wd + (size_t)colB0 * K + k0 + 32, 0, 3);
    }
    v16bf a0 = load_a_frag(Ad, K, rowA0, k0, half);
    v16bf a1 = load_a_frag(Ad, K, rowA1, k0, half);
    v16bf b0 = load_b_frag(Wd, K, colB0, k0, half);
    v16bf b1 = load_b_frag(Wd, K, colB1, k0, half);
    acc00 = __builtin_amdgcn_wmma_f32_16x16x32_bf16(false, a0, false, b0, (short)0, acc00, false, false);
    acc01 = __builtin_amdgcn_wmma_f32_16x16x32_bf16(false, a0, false, b1, (short)0, acc01, false, false);
    acc10 = __builtin_amdgcn_wmma_f32_16x16x32_bf16(false, a1, false, b0, (short)0, acc10, false, false);
    acc11 = __builtin_amdgcn_wmma_f32_16x16x32_bf16(false, a1, false, b1, (short)0, acc11, false, false);
  }

  float bv0 = 0.0f, bv1 = 0.0f;
  if (bias) {
    const float* bd = bias + (size_t)d * sBd;
    bv0 = bd[colB0];
    bv1 = bd[colB1];
  }

  // C/D layout: VGPR r -> row = half*8 + r ; col = lane&15 (per n-tile).
#pragma unroll
  for (int r = 0; r < 8; ++r) {
    const int row0 = m0 + half * 8 + r;
    const int row1 = row0 + 16;
    if (OUT_BF16) {
      unsigned short* O_ = (unsigned short*)Out + (size_t)d * sOd;
      O_[(size_t)row0 * N + colB0] = f2bf(acc00[r] + bv0);
      O_[(size_t)row0 * N + colB1] = f2bf(acc01[r] + bv1);
      O_[(size_t)row1 * N + colB0] = f2bf(acc10[r] + bv0);
      O_[(size_t)row1 * N + colB1] = f2bf(acc11[r] + bv1);
    } else {
      float* O_ = (float*)Out + (size_t)d * sOd;
      O_[(size_t)row0 * N + colB0] = acc00[r] + bv0;
      O_[(size_t)row0 * N + colB1] = acc01[r] + bv1;
      O_[(size_t)row1 * N + colB0] = acc10[r] + bv0;
      O_[(size_t)row1 * N + colB1] = acc11[r] + bv1;
    }
  }
}

// ---------------------------------------------------------------------------
// Column batch-stats (mean, rstd) over X viewed as [Dn][rows][On]; column id
// c = d*On + o matches BatchNorm over reshape(B, Dn*On).
// ---------------------------------------------------------------------------
__global__ void colstats_f32(const float* __restrict__ X, int Dn, int rows, int On,
                             float* __restrict__ mean, float* __restrict__ rstd) {
  int c = blockIdx.x * blockDim.x + threadIdx.x;
  if (c >= Dn * On) return;
  int d = c / On, o = c % On;
  const float* p = X + (size_t)d * rows * On + o;
  float s = 0.f, s2 = 0.f;
  for (int r = 0; r < rows; ++r) { float v = p[(size_t)r * On]; s += v; s2 += v * v; }
  float m = s / rows;
  float var = s2 / rows - m * m;
  mean[c] = m;
  rstd[c] = rsqrtf(var + EPSQ);
}

__global__ void colstats_bf16(const unsigned short* __restrict__ X, int Dn, int rows, int On,
                              float* __restrict__ mean, float* __restrict__ rstd) {
  int c = blockIdx.x * blockDim.x + threadIdx.x;
  if (c >= Dn * On) return;
  int d = c / On, o = c % On;
  const unsigned short* p = X + (size_t)d * rows * On + o;
  float s = 0.f, s2 = 0.f;
  for (int r = 0; r < rows; ++r) { float v = bf2f(p[(size_t)r * On]); s += v; s2 += v * v; }
  float m = s / rows;
  float var = s2 / rows - m * m;
  mean[c] = m;
  rstd[c] = rsqrtf(var + EPSQ);
}

// xn[b,f] = (x[b,f]-mean1[f])*rstd1[f]  -> bf16   (gamma/beta folded into W/bias)
__global__ void normx_bf16(const float* __restrict__ x, const float* __restrict__ mean,
                           const float* __restrict__ rstd, unsigned short* __restrict__ xn) {
  long long i = (long long)blockIdx.x * blockDim.x + threadIdx.x;
  if (i >= (long long)BQ * FQ) return;
  int f = (int)(i % FQ);
  xn[i] = f2bf((x[i] - mean[f]) * rstd[f]);
}

// Wb1[d,o,f] = bf16(W1[d,o,f] * g1[d*F+f])
__global__ void w1scale_bf16(const float* __restrict__ W1, const float* __restrict__ g1,
                             unsigned short* __restrict__ Wb1) {
  long long i = (long long)blockIdx.x * blockDim.x + threadIdx.x;
  if (i >= (long long)DQ * FQ * FQ) return;
  int d = (int)(i / ((long long)FQ * FQ));
  int f = (int)(i % FQ);
  Wb1[i] = f2bf(W1[i] * g1[d * FQ + f]);
}

// bias1[d,o] = sum_f b1[d*F+f] * W1[d,o,f]   (one block per (d,o))
__global__ void bias1_kernel(const float* __restrict__ W1, const float* __restrict__ b1,
                             float* __restrict__ bias1) {
  int d = blockIdx.x / FQ, o = blockIdx.x % FQ;
  const float* wrow = W1 + ((size_t)d * FQ + o) * FQ;
  const float* brow = b1 + (size_t)d * FQ;
  float s = 0.f;
  for (int f = threadIdx.x; f < FQ; f += blockDim.x) s += wrow[f] * brow[f];
  __shared__ float red[256];
  red[threadIdx.x] = s;
  __syncthreads();
  for (int st = 128; st > 0; st >>= 1) {
    if ((int)threadIdx.x < st) red[threadIdx.x] += red[threadIdx.x + st];
    __syncthreads();
  }
  if (threadIdx.x == 0) bias1[blockIdx.x] = red[0];
}

__global__ void f32_to_bf16_copy(const float* __restrict__ src, unsigned short* __restrict__ dst,
                                 long long n) {
  long long i = (long long)blockIdx.x * blockDim.x + threadIdx.x;
  if (i < n) dst[i] = f2bf(src[i]);
}

// In-place over h[d][rows][O] bf16: h = sigmoid((h-mean)*rstd*g + b), col = d*O+o.
__global__ void bnsig_bf16(unsigned short* __restrict__ h, const float* __restrict__ mean,
                           const float* __restrict__ rstd, const float* __restrict__ g,
                           const float* __restrict__ b) {
  long long i = (long long)blockIdx.x * blockDim.x + threadIdx.x;
  if (i >= (long long)DQ * BQ * FQ) return;
  int o = (int)(i % FQ);
  int d = (int)(i / ((long long)BQ * FQ));
  int c = d * FQ + o;
  float v = bf2f(h[i]);
  v = (v - mean[c]) * rstd[c] * g[c] + b[c];
  h[i] = f2bf(sigmoidf_(v));
}

// Final: y[b, d*32+o] = sigmoid((out2[d][b,o]-mean)*rstd*g + b)
__global__ void bnsig_final(const float* __restrict__ out2, const float* __restrict__ mean,
                            const float* __restrict__ rstd, const float* __restrict__ g,
                            const float* __restrict__ b, float* __restrict__ y) {
  long long i = (long long)blockIdx.x * blockDim.x + threadIdx.x;
  if (i >= (long long)DQ * BQ * O2Q) return;
  int o = (int)(i % O2Q);
  long long t = i / O2Q;
  int r = (int)(t % BQ);
  int d = (int)(t / BQ);
  int c = d * O2Q + o;
  float v = out2[i];
  y[(size_t)r * (DQ * O2Q) + c] = sigmoidf_((v - mean[c]) * rstd[c] * g[c] + b[c]);
}

// ---------------------------------------------------------------------------
// Host-side launcher
// ---------------------------------------------------------------------------
extern "C" void kernel_launch(void* const* d_in, const int* in_sizes, int n_in,
                              void* d_out, int out_size, void* d_ws, size_t ws_size,
                              hipStream_t stream) {
  (void)in_sizes; (void)n_in; (void)out_size; (void)ws_size;

  const float* x  = (const float*)d_in[0];
  const float* W1 = (const float*)d_in[1];
  const float* W2 = (const float*)d_in[2];
  const float* g1 = (const float*)d_in[3];
  const float* b1 = (const float*)d_in[4];
  const float* g2 = (const float*)d_in[5];
  const float* b2 = (const float*)d_in[6];
  const float* g3 = (const float*)d_in[7];
  const float* b3 = (const float*)d_in[8];
  (void)g1; (void)b1; // used below (folded)

  // Workspace carve-out (256-byte aligned regions).
  char* base = (char*)d_ws;
  auto carve = [&](size_t bytes) -> char* {
    char* p = base;
    base += (bytes + 255) & ~(size_t)255;
    return p;
  };
  float* mean1 = (float*)carve(FQ * sizeof(float));
  float* rstd1 = (float*)carve(FQ * sizeof(float));
  float* mean2 = (float*)carve((size_t)DQ * FQ * sizeof(float));
  float* rstd2 = (float*)carve((size_t)DQ * FQ * sizeof(float));
  float* mean3 = (float*)carve(DQ * O2Q * sizeof(float));
  float* rstd3 = (float*)carve(DQ * O2Q * sizeof(float));
  float* bias1 = (float*)carve((size_t)DQ * FQ * sizeof(float));
  unsigned short* xn   = (unsigned short*)carve((size_t)BQ * FQ * 2);          // 33.5 MB
  unsigned short* Wb1  = (unsigned short*)carve((size_t)DQ * FQ * FQ * 2);     // 8.4 MB
  unsigned short* Wb2  = (unsigned short*)carve((size_t)DQ * O2Q * FQ * 2);    // 0.26 MB
  unsigned short* out1 = (unsigned short*)carve((size_t)DQ * BQ * FQ * 2);     // 134 MB
  float*          out2 = (float*)carve((size_t)DQ * BQ * O2Q * sizeof(float)); // 8.4 MB

  // 1) BN1 stats on x (identical for all D replicas of the concat).
  colstats_f32<<<(FQ + 255) / 256, 256, 0, stream>>>(x, 1, BQ, FQ, mean1, rstd1);

  // 2) Normalize x -> bf16.
  normx_bf16<<<((long long)BQ * FQ) / 256, 256, 0, stream>>>(x, mean1, rstd1, xn);

  // 3) Fold gamma1 into W1 (bf16); bias1 = b1 . W1 ; W2 -> bf16.
  w1scale_bf16<<<((long long)DQ * FQ * FQ) / 256, 256, 0, stream>>>(W1, g1, Wb1);
  bias1_kernel<<<DQ * FQ, 256, 0, stream>>>(W1, b1, bias1);
  f32_to_bf16_copy<<<((long long)DQ * O2Q * FQ + 255) / 256, 256, 0, stream>>>(
      W2, Wb2, (long long)DQ * O2Q * FQ);

  // 4) GEMM1: out1[d] = xn * (g1.W1[d])^T + bias1[d]   (bf16 out, f32 accum)
  gemm_wmma_bf16<true><<<dim3(BQ / 128, FQ / 32, DQ), 128, 0, stream>>>(
      xn, Wb1, bias1, out1, BQ, FQ, FQ,
      /*sAd=*/0, /*sWd=*/(size_t)FQ * FQ, /*sOd=*/(size_t)BQ * FQ, /*sBd=*/(size_t)FQ);

  // 5) BN2 stats + sigmoid(BN2) in place.
  colstats_bf16<<<(DQ * FQ + 255) / 256, 256, 0, stream>>>(out1, DQ, BQ, FQ, mean2, rstd2);
  bnsig_bf16<<<((long long)DQ * BQ * FQ) / 256, 256, 0, stream>>>(out1, mean2, rstd2, g2, b2);

  // 6) GEMM2: out2[d] = s2 * W2[d]^T   (f32 out)
  gemm_wmma_bf16<false><<<dim3(BQ / 128, O2Q / 32, DQ), 128, 0, stream>>>(
      out1, Wb2, nullptr, out2, BQ, O2Q, FQ,
      /*sAd=*/(size_t)BQ * FQ, /*sWd=*/(size_t)O2Q * FQ, /*sOd=*/(size_t)BQ * O2Q, /*sBd=*/0);

  // 7) BN3 stats + final sigmoid into d_out [B, D*32].
  colstats_f32<<<(DQ * O2Q + 255) / 256, 256, 0, stream>>>(out2, DQ, BQ, O2Q, mean3, rstd3);
  bnsig_final<<<((long long)DQ * BQ * O2Q) / 256, 256, 0, stream>>>(
      out2, mean3, rstd3, g3, b3, (float*)d_out);
}
